// TransPITF_1211180777751
// MI455X (gfx1250) — compile-verified, matrix operand order checked by reading, not run on hardware
//
#include <hip/hip_runtime.h>

typedef __attribute__((ext_vector_type(16))) __bf16 v16bf;
typedef __attribute__((ext_vector_type(8)))  float  v8f;

#define KDIM 64
#define ROWS_PER_WAVE 16
#define ROWS_PER_BLOCK 128   // 8 waves * 16 rows
#define LDS_STRIDE 72        // shorts per row (144 B, 16B-aligned, bank-spread)

union FragBF {
  v16bf  v;
  unsigned u[8];
  uint4  q[2];
};

// Truncating f32->bf16 pair pack: {bf16(b), bf16(a)} in one dword.
// v_perm_b32 selects bytes {7,6,3,2} of {S0=b, S1=a}.
__device__ __forceinline__ unsigned pk_bf16(float a, float b) {
#if __has_builtin(__builtin_amdgcn_perm)
  return __builtin_amdgcn_perm(__float_as_uint(b), __float_as_uint(a), 0x07060302u);
#else
  return (__float_as_uint(a) >> 16) | (__float_as_uint(b) & 0xffff0000u);
#endif
}

__device__ __forceinline__ float fast_tanh(float x) {
#if __has_builtin(__builtin_amdgcn_tanhf)
  return __builtin_amdgcn_tanhf(x);
#elif __has_builtin(__builtin_amdgcn_tanh_f32)
  return __builtin_amdgcn_tanh_f32(x);
#else
  // tanh(x) = 2*sigmoid(2x) - 1
  float e = __expf(-2.0f * x);
  return 2.0f * __builtin_amdgcn_rcpf(1.0f + e) - 1.0f;
#endif
}

__global__ __launch_bounds__(256)
void pitf_wmma_kernel(const int*   __restrict__ x,
                      const float* __restrict__ user_vecs,
                      const float* __restrict__ item_vecs,
                      const float* __restrict__ tag_vecs,
                      const float* __restrict__ w_user,
                      const float* __restrict__ b_user,
                      const float* __restrict__ w_item,
                      const float* __restrict__ b_item,
                      float*       __restrict__ out,
                      int nrows)
{
  __shared__ unsigned short lw[2][64][LDS_STRIDE];

  // Stage both 64x64 weight matrices into LDS as bf16, PRE-SCALED by 0.5:
  // sig(x)-sig(y) = 0.5*(tanh(x/2)-tanh(y/2)); the /2 folds into W and b.
  // 4096 dwords total: [2][64][32] bf16 pairs.
  for (int d = threadIdx.x; d < 2 * 64 * 32; d += 256) {
    int m   = d >> 11;        // 0 = w_user, 1 = w_item
    int rem = d & 2047;
    int n  = rem >> 5;
    int kk = rem & 31;        // dword (pair of k) within row
    const float* wsrc = (m ? w_item : w_user) + n * 64 + kk * 2;
    float2 wv = *(const float2*)wsrc;
    *(unsigned*)&lw[m][n][kk * 2] = pk_bf16(0.5f * wv.x, 0.5f * wv.y);
  }
  __syncthreads();

  const int lane = threadIdx.x & 31;
  const int wid  = threadIdx.x >> 5;
  const int rowbase = blockIdx.x * ROWS_PER_BLOCK + wid * ROWS_PER_WAVE;
  if (rowbase + ROWS_PER_WAVE > nrows) return;   // uniform per wave

  const int hl   = lane & 15;     // position in half-wave
  const int hi   = lane >> 4;     // half-wave select
  const int koff = hi ? 8 : 0;    // K-run base for 16-bit A/B fragment layout

  // ---- gather indices (one b128 per row) ----
  const int arow = rowbase + hl;               // lane -> M for A fragments
  const int4 qa = *(const int4*)(x + arow * 4);
  const int tidx = qa.z;
  const int nidx = qa.w;
  const float* tagp = tag_vecs + (size_t)tidx * KDIM;
  const float* negp = tag_vecs + (size_t)nidx * KDIM;

  // 32-bit byte offsets for user/item gathers: SGPR base + VGPR offset + imm(nt*64)
  unsigned uoff[8], ioff[8];
#pragma unroll
  for (int v = 0; v < 8; ++v) {
    int r = rowbase + v + hi * 8;              // C/D layout: vgpr v -> row v (+8 upper half)
    const int4 q = *(const int4*)(x + r * 4);
    uoff[v] = ((unsigned)q.x * KDIM + (unsigned)hl) * 4u;
    ioff[v] = ((unsigned)q.y * KDIM + (unsigned)hl) * 4u;
  }
#pragma unroll
  for (int v = 0; v < 8; ++v) {
    __builtin_prefetch((const char*)user_vecs + uoff[v], 0, 0);
    __builtin_prefetch((const char*)item_vecs + ioff[v], 0, 0);
  }

  // ---- build bf16 A fragments (tag, neg_tag), 2 chunks of K=32 each ----
  FragBF tagA[2], negA[2];
#pragma unroll
  for (int c = 0; c < 2; ++c) {
    const float* tp = tagp + c * 32 + koff;
    const float* np = negp + c * 32 + koff;
    float4 t0 = *(const float4*)(tp);
    float4 t1 = *(const float4*)(tp + 4);
    float4 t2 = *(const float4*)(tp + 16);
    float4 t3 = *(const float4*)(tp + 20);
    tagA[c].u[0] = pk_bf16(t0.x, t0.y); tagA[c].u[1] = pk_bf16(t0.z, t0.w);
    tagA[c].u[2] = pk_bf16(t1.x, t1.y); tagA[c].u[3] = pk_bf16(t1.z, t1.w);
    tagA[c].u[4] = pk_bf16(t2.x, t2.y); tagA[c].u[5] = pk_bf16(t2.z, t2.w);
    tagA[c].u[6] = pk_bf16(t3.x, t3.y); tagA[c].u[7] = pk_bf16(t3.z, t3.w);
    float4 m0 = *(const float4*)(np);
    float4 m1 = *(const float4*)(np + 4);
    float4 m2 = *(const float4*)(np + 16);
    float4 m3 = *(const float4*)(np + 20);
    negA[c].u[0] = pk_bf16(m0.x, m0.y); negA[c].u[1] = pk_bf16(m0.z, m0.w);
    negA[c].u[2] = pk_bf16(m1.x, m1.y); negA[c].u[3] = pk_bf16(m1.z, m1.w);
    negA[c].u[4] = pk_bf16(m2.x, m2.y); negA[c].u[5] = pk_bf16(m2.z, m2.w);
    negA[c].u[6] = pk_bf16(m3.x, m3.y); negA[c].u[7] = pk_bf16(m3.z, m3.w);
  }

  float racc[8];
#pragma unroll
  for (int v = 0; v < 8; ++v) racc[v] = 0.0f;

  // ---- 4 N-tiles of 16 output features each ----
#pragma unroll
  for (int nt = 0; nt < 4; ++nt) {
    const int n = nt * 16 + hl;

    // B fragments for 0.5*w_user / 0.5*w_item from LDS (element (K=k,N=n) = w[n][k]/2)
    FragBF wu[2], wi[2];
#pragma unroll
    for (int c = 0; c < 2; ++c) {
      const unsigned short* pu = &lw[0][n][c * 32 + koff];
      wu[c].q[0] = *(const uint4*)(pu);
      wu[c].q[1] = *(const uint4*)(pu + 16);
      const unsigned short* pi = &lw[1][n][c * 32 + koff];
      wi[c].q[0] = *(const uint4*)(pi);
      wi[c].q[1] = *(const uint4*)(pi + 16);
    }

    // bias/2 broadcast into the WMMA C operand (column-constant => all 8 regs equal)
    const float bu05 = 0.5f * b_user[n];
    const float bi05 = 0.5f * b_item[n];
    v8f bu8, bi8;
#pragma unroll
    for (int v = 0; v < 8; ++v) { bu8[v] = bu05; bi8[v] = bi05; }

    // user/item elements in C/D fragment layout for this N-tile (imm offset = nt*64B)
    float uf[8], itf[8];
#pragma unroll
    for (int v = 0; v < 8; ++v) {
      uf[v]  = *(const float*)((const char*)user_vecs + uoff[v] + nt * 64);
      itf[v] = *(const float*)((const char*)item_vecs + ioff[v] + nt * 64);
    }

    // Four independent WMMA chains: chunk0 of all four, then chunk1 of all four.
    // Keeps >=3 instructions between each D->C chain link and defers all D->VALU
    // reads past 4 subsequent WMMAs (fills the bf16 WMMA hazard slots).
    v8f auU, anU, auI, anI;
    auU = __builtin_amdgcn_wmma_f32_16x16x32_bf16(false, tagA[0].v, false, wu[0].v,
                                                  (short)0, bu8, false, false);
    anU = __builtin_amdgcn_wmma_f32_16x16x32_bf16(false, negA[0].v, false, wu[0].v,
                                                  (short)0, bu8, false, false);
    auI = __builtin_amdgcn_wmma_f32_16x16x32_bf16(false, tagA[0].v, false, wi[0].v,
                                                  (short)0, bi8, false, false);
    anI = __builtin_amdgcn_wmma_f32_16x16x32_bf16(false, negA[0].v, false, wi[0].v,
                                                  (short)0, bi8, false, false);
    auU = __builtin_amdgcn_wmma_f32_16x16x32_bf16(false, tagA[1].v, false, wu[1].v,
                                                  (short)0, auU, false, false);
    anU = __builtin_amdgcn_wmma_f32_16x16x32_bf16(false, negA[1].v, false, wu[1].v,
                                                  (short)0, anU, false, false);
    auI = __builtin_amdgcn_wmma_f32_16x16x32_bf16(false, tagA[1].v, false, wi[1].v,
                                                  (short)0, auI, false, false);
    anI = __builtin_amdgcn_wmma_f32_16x16x32_bf16(false, negA[1].v, false, wi[1].v,
                                                  (short)0, anI, false, false);

#pragma unroll
    for (int v = 0; v < 8; ++v) {
      float du = fast_tanh(auU[v]) - fast_tanh(anU[v]);
      float di = fast_tanh(auI[v]) - fast_tanh(anI[v]);
      racc[v] = fmaf(uf[v], du, racc[v]);
      racc[v] = fmaf(itf[v], di, racc[v]);
    }
  }

  // ---- reduce over N within each half-wave (16 lanes), apply global 0.5 ----
#pragma unroll
  for (int v = 0; v < 8; ++v) {
    float s = racc[v];
    s += __shfl_xor(s, 1, 32);
    s += __shfl_xor(s, 2, 32);
    s += __shfl_xor(s, 4, 32);
    s += __shfl_xor(s, 8, 32);
    racc[v] = 0.5f * s;
  }

  if (hl == 0) {
#pragma unroll
    for (int v = 0; v < 8; ++v) out[rowbase + hi * 8 + v] = racc[v];
  }
}

extern "C" void kernel_launch(void* const* d_in, const int* in_sizes, int n_in,
                              void* d_out, int out_size, void* d_ws, size_t ws_size,
                              hipStream_t stream) {
  const int*   x         = (const int*)  d_in[0];
  const float* user_vecs = (const float*)d_in[1];
  const float* item_vecs = (const float*)d_in[2];
  const float* tag_vecs  = (const float*)d_in[3];
  const float* w_user    = (const float*)d_in[4];
  const float* b_user    = (const float*)d_in[5];
  const float* w_item    = (const float*)d_in[6];
  const float* b_item    = (const float*)d_in[7];
  float* out = (float*)d_out;

  const int nrows = in_sizes[0] / 4;                 // B
  const int nblocks = (nrows + ROWS_PER_BLOCK - 1) / ROWS_PER_BLOCK;

  pitf_wmma_kernel<<<nblocks, 256, 0, stream>>>(
      x, user_vecs, item_vecs, tag_vecs, w_user, b_user, w_item, b_item,
      out, nrows);
}
